// FeatureAlignmentLayer_15298673509080
// MI455X (gfx1250) — matrix operand order repaired and checked by reading, not compiled, not run on hardware
//
#include <hip/hip_runtime.h>
#include <hip/hip_bf16.h>

// ---------------------------------------------------------------------------
// Types / helpers
// ---------------------------------------------------------------------------
typedef __attribute__((ext_vector_type(16))) __bf16 v16bf;
typedef __attribute__((ext_vector_type(8)))  float  v8f;

union FragBits { unsigned int u[8]; v16bf v; };

__device__ __forceinline__ unsigned short f2bf(float x) {
    unsigned int u = __float_as_uint(x);
    u += 0x7fffu + ((u >> 16) & 1u);          // round-to-nearest-even
    return (unsigned short)(u >> 16);
}
// pack two floats into two bf16 in one dword (lo = a, hi = b)
__device__ __forceinline__ unsigned int pack2bf(float a, float b) {
    unsigned int ua = __float_as_uint(a);
    unsigned int ub = __float_as_uint(b);
    ua += 0x7fffu + ((ua >> 16) & 1u);
    ub += 0x7fffu + ((ub >> 16) & 1u);
    return (ua >> 16) | (ub & 0xffff0000u);
}
__device__ __forceinline__ float gelu_f(float x) {
    return 0.5f * x * (1.0f + erff(x * 0.70710678118654752f)); // exact erf GELU
}
__device__ __forceinline__ v8f wmma_bf16(v16bf a, v16bf b, v8f c) {
    return __builtin_amdgcn_wmma_f32_16x16x32_bf16(false, a, false, b,
                                                   (short)0, c, false, false);
}

// A fragment: 16x32 bf16, LDS layout lds[m][k], k contiguous, given stride.
// lane<16: K pairs {0,2,4,6,16,18,20,22}; lane>=16: +8 (per CDNA5 ISA layout)
__device__ __forceinline__ v16bf load_fragA(const unsigned short* lds, int stride, int lane) {
    int m  = lane & 15;
    int hi = (lane >> 4) & 1;
    const unsigned short* p = lds + m * stride + hi * 8;
    FragBits f;
#pragma unroll
    for (int i = 0; i < 8; ++i) {
        int k = (i & 3) * 2 + (i >> 2) * 16;
        f.u[i] = *(const unsigned int*)(p + k);
    }
    return f.v;
}
// B fragment: 32x16 bf16, LDS layout lds[n][k] (transposed so K pairs are
// contiguous). lane<16: K pairs {0,2,4,6,16,18,20,22}; lane>=16: +8.
__device__ __forceinline__ v16bf load_fragB(const unsigned short* lds, int stride, int lane) {
    int n  = lane & 15;
    int hi = (lane >> 4) & 1;
    const unsigned short* p = lds + n * stride + hi * 8;
    FragBits f;
#pragma unroll
    for (int i = 0; i < 8; ++i) f.u[i] = *(const unsigned int*)(p + (i & 3) * 2 + (i >> 2) * 16);
    return f.v;
}

#define LG 1024  // sequence length L

// ---------------------------------------------------------------------------
// Elementwise GELU
// ---------------------------------------------------------------------------
__global__ __launch_bounds__(256) void k_gelu(const float* __restrict__ x,
                                              float* __restrict__ y, int n) {
    int i = blockIdx.x * 256 + threadIdx.x;
    if (i < n) y[i] = gelu_f(x[i]);
}

// ---------------------------------------------------------------------------
// GEMM: Y[n] = W(Cout x Cin) @ X[n](Cin x L), bf16 WMMA, f32 accumulate.
// Tiles: 64 (M) x 128 (L) x 32 (K). Block = 256 threads = 8 waves.
// ---------------------------------------------------------------------------
__global__ __launch_bounds__(256) void k_gemm(const float* __restrict__ X,
                                              const float* __restrict__ W,
                                              float* __restrict__ Y,
                                              int Cin, int Cout) {
    __shared__ unsigned short As[64 * 34];    // W tile  [m][k]
    __shared__ unsigned short Bs[128 * 34];   // X tile  [l][k] (transposed)
    const int t = threadIdx.x, lane = t & 31, wave = t >> 5;
    const int mrow = wave & 3, nh = wave >> 2;
    const int n = blockIdx.z;
    const int l0 = blockIdx.x * 128, m0 = blockIdx.y * 64;
    const float* Xn = X + (size_t)n * Cin * LG;

    v8f acc[4];
#pragma unroll
    for (int j = 0; j < 4; ++j)
#pragma unroll
        for (int i = 0; i < 8; ++i) acc[j][i] = 0.0f;

    for (int k0 = 0; k0 < Cin; k0 += 32) {
        __syncthreads();
        {   // W tile 64x32: thread -> row t/4, 8 cols (float2 loads, packed stores)
            int row = t >> 2, cb = (t & 3) * 8;
            const float2* wp = (const float2*)(W + (size_t)(m0 + row) * Cin + k0 + cb);
            unsigned int* ap = (unsigned int*)(As + row * 34 + cb);
#pragma unroll
            for (int i = 0; i < 4; ++i) {
                float2 w2 = wp[i];
                ap[i] = pack2bf(w2.x, w2.y);
            }
        }
        // X tile 32x128 -> Bs[l][k]; two K rows per thread -> one packed store
#pragma unroll
        for (int rep = 0; rep < 8; ++rep) {
            int idx = rep * 256 + t;           // 2048 pair jobs
            int kk = (idx >> 7) * 2, ll = idx & 127;
            float x0 = Xn[(size_t)(k0 + kk) * LG + l0 + ll];
            float x1 = Xn[(size_t)(k0 + kk + 1) * LG + l0 + ll];
            *(unsigned int*)&Bs[ll * 34 + kk] = pack2bf(x0, x1);
        }
        if (k0 + 32 < Cin) {  // prefetch next X tile (global_prefetch_b8)
            __builtin_prefetch(Xn + (size_t)(k0 + 32 + (t >> 3)) * LG + l0 + (t & 7) * 16, 0, 3);
        }
        __syncthreads();
        v16bf a = load_fragA(As + (mrow * 16) * 34, 34, lane);
#pragma unroll
        for (int j = 0; j < 4; ++j) {
            v16bf b = load_fragB(Bs + (nh * 64 + j * 16) * 34, 34, lane);
            acc[j] = wmma_bf16(a, b, acc[j]);
        }
    }
    // write back: lane<16 -> rows r, lane>=16 -> rows r+8
    const int rbase = m0 + mrow * 16 + ((lane >> 4) ? 8 : 0);
#pragma unroll
    for (int j = 0; j < 4; ++j) {
        int col = l0 + nh * 64 + j * 16 + (lane & 15);
#pragma unroll
        for (int r = 0; r < 8; ++r)
            Y[((size_t)n * Cout + rbase + r) * LG + col] = acc[j][r];
    }
}

// ---------------------------------------------------------------------------
// Per-sample mean / rsqrt(var+eps) over C*L elements (GroupNorm(1) / LN[-2,-1])
// ---------------------------------------------------------------------------
__global__ __launch_bounds__(256) void k_stats(const float* __restrict__ x,
                                               float* __restrict__ mu,
                                               float* __restrict__ rs, int perN) {
    __shared__ float s1[256], s2[256];
    const int n = blockIdx.x;
    const float* p = x + (size_t)n * perN;
    float a = 0.f, b = 0.f;
    for (int i = threadIdx.x; i < perN; i += 256) { float v = p[i]; a += v; b += v * v; }
    s1[threadIdx.x] = a; s2[threadIdx.x] = b;
    __syncthreads();
    for (int s = 128; s > 0; s >>= 1) {
        if (threadIdx.x < s) { s1[threadIdx.x] += s1[threadIdx.x + s]; s2[threadIdx.x] += s2[threadIdx.x + s]; }
        __syncthreads();
    }
    if (threadIdx.x == 0) {
        float m = s1[0] / perN;
        float v = s2[0] / perN - m * m;
        mu[n] = m; rs[n] = rsqrtf(v + 1e-5f);
    }
}

// y = [gelu]( [resid +] (x-mu)*rs [*w[c]+b[c]] )
__global__ __launch_bounds__(256) void k_gn_apply(const float* __restrict__ x,
                                                  const float* __restrict__ resid,
                                                  const float* __restrict__ w,
                                                  const float* __restrict__ b,
                                                  const float* __restrict__ mu,
                                                  const float* __restrict__ rs,
                                                  float* __restrict__ y,
                                                  int C, int do_gelu) {
    int idx = blockIdx.x * 256 + threadIdx.x;
    int perN = C * LG;
    int n = idx / perN, c = (idx / LG) % C;
    float v = (x[idx] - mu[n]) * rs[n];
    if (w) v = v * w[c] + b[c];
    if (resid) v += resid[idx];
    if (do_gelu) v = gelu_f(v);
    y[idx] = v;
}

// ---------------------------------------------------------------------------
// Attention LN statistics without materializing logits:
//   sum(S)  = sum_c (sum_l q[c,l]) (sum_m k[c,m])
//   sum(S²) = sum_{c,c'} Qcov[c,c'] Kcov[c,c']   (Qcov = Q Qᵀ, 16x16)
// One block per (n,h).
// ---------------------------------------------------------------------------
__global__ __launch_bounds__(256) void k_attn_stats(const float* __restrict__ qkv,
                                                    float* __restrict__ muH,
                                                    float* __restrict__ rsH) {
    __shared__ float Qc[136], Kc[136], sq[16], sk[16];
    const int nh = blockIdx.x, n = nh >> 3, h = nh & 7;
    const float* q = qkv + ((size_t)n * 512 + h * 16) * LG;
    const float* k = qkv + ((size_t)n * 512 + 128 + h * 16) * LG;
    for (int job = threadIdx.x; job < 304; job += 256) {
        if (job < 272) {
            int p = (job < 136) ? job : job - 136;
            const float* base = (job < 136) ? q : k;
            int i = 0;
            while (p >= 16 - i) { p -= 16 - i; ++i; }
            int jj = i + p;
            const float* ri = base + (size_t)i * LG;
            const float* rj = base + (size_t)jj * LG;
            float s = 0.f;
            for (int l = 0; l < LG; ++l) s += ri[l] * rj[l];
            if (job < 136) Qc[job] = s; else Kc[job - 136] = s;
        } else if (job < 288) {
            int c = job - 272; const float* r = q + (size_t)c * LG;
            float s = 0.f; for (int l = 0; l < LG; ++l) s += r[l];
            sq[c] = s;
        } else {
            int c = job - 288; const float* r = k + (size_t)c * LG;
            float s = 0.f; for (int l = 0; l < LG; ++l) s += r[l];
            sk[c] = s;
        }
    }
    __syncthreads();
    if (threadIdx.x == 0) {
        float sumS = 0.f;
        for (int c = 0; c < 16; ++c) sumS += sq[c] * sk[c];
        const float inv = 1.0f / ((float)LG * (float)LG);
        float mean = sumS * inv;
        float ss = 0.f; int p = 0;
        for (int i = 0; i < 16; ++i)
            for (int j = i; j < 16; ++j, ++p)
                ss += ((i == j) ? 1.0f : 2.0f) * Qc[p] * Kc[p];
        float var = ss * inv - mean * mean;
        muH[nh] = mean; rsH[nh] = rsqrtf(var + 1e-5f);
    }
}

// ---------------------------------------------------------------------------
// Fused attention (flash style): per block = 64 rows of one (n,h).
// S tiles via bf16 WMMA (K=16 zero-padded to 32), LN affine + diag mask +
// exp in-register, P@Vᵀ accumulated with bf16 WMMA, row sums in LDS,
// final 1/rowsum applied on writeback. Logits never touch HBM.
// ---------------------------------------------------------------------------
__global__ __launch_bounds__(256) void k_attn_fused(const float* __restrict__ qkv,
                                                    const float* __restrict__ muH,
                                                    const float* __restrict__ rsH,
                                                    float* __restrict__ out) {
    __shared__ unsigned short Aq[64 * 34];    // qᵀ tile [l][c], c16..31 zero
    __shared__ unsigned short Ks[128 * 34];   // k tile  [m][c], c16..31 zero
    __shared__ unsigned short Vs[32 * 130];   // v tile  [d][m]
    __shared__ unsigned short Ps[64 * 130];   // P tile  [l][m] (bf16 probs)
    __shared__ float Ored[64 * 33];
    __shared__ float rowsum[64];

    const int t = threadIdx.x, lane = t & 31, wave = t >> 5;
    const int mrow = wave & 3, nh = wave >> 2;
    const int nhid = blockIdx.y, n = nhid >> 3, h = nhid & 7;
    const int l0 = blockIdx.x * 64;
    const float* q = qkv + ((size_t)n * 512 + h * 16) * LG;
    const float* k = qkv + ((size_t)n * 512 + 128 + h * 16) * LG;
    const float* v = qkv + ((size_t)n * 512 + 256 + h * 32) * LG;
    const float mu = muH[nhid], rsg = rsH[nhid];

    for (int i = t; i < 64 * 17; i += 256)  *(unsigned int*)&Aq[i * 2] = 0u;
    for (int i = t; i < 128 * 17; i += 256) *(unsigned int*)&Ks[i * 2] = 0u;
    if (t < 64) rowsum[t] = 0.0f;
    __syncthreads();
    for (int i = t; i < 64 * 8; i += 256) {          // q tile, packed c pairs
        int l = i >> 3, c = (i & 7) * 2;
        *(unsigned int*)&Aq[l * 34 + c] =
            pack2bf(q[(size_t)c * LG + l0 + l], q[(size_t)(c + 1) * LG + l0 + l]);
    }

    v8f oacc[2];
#pragma unroll
    for (int j = 0; j < 2; ++j)
#pragma unroll
        for (int i = 0; i < 8; ++i) oacc[j][i] = 0.0f;

    for (int mt = 0; mt < 8; ++mt) {
        const int m0 = mt * 128;
        __syncthreads();
#pragma unroll
        for (int rep = 0; rep < 4; ++rep) {          // Ks[m][c], packed c pairs
            int idx = rep * 256 + t;                  // 1024 pair jobs
            int c = (idx >> 7) * 2, m = idx & 127;
            *(unsigned int*)&Ks[m * 34 + c] =
                pack2bf(k[(size_t)c * LG + m0 + m], k[(size_t)(c + 1) * LG + m0 + m]);
        }
#pragma unroll
        for (int rep = 0; rep < 8; ++rep) {          // Vs[d][m], packed m pairs
            int idx = rep * 256 + t;                  // 2048 pair jobs
            int d = idx >> 6, mp = (idx & 63) * 2;
            float2 v2 = *(const float2*)(v + (size_t)d * LG + m0 + mp);
            *(unsigned int*)&Vs[d * 130 + mp] = pack2bf(v2.x, v2.y);
        }
        __syncthreads();

        // ---- S = qᵀk tiles, LN affine, diag mask, exp -> Ps ----
        v16bf afr = load_fragA(Aq + (mrow * 16) * 34, 34, lane);
#pragma unroll
        for (int j = 0; j < 4; ++j) {
            v16bf bfr = load_fragB(Ks + (nh * 64 + j * 16) * 34, 34, lane);
            v8f sacc;
#pragma unroll
            for (int i = 0; i < 8; ++i) sacc[i] = 0.0f;
            sacc = wmma_bf16(afr, bfr, sacc);
            const int colL = nh * 64 + j * 16 + (lane & 15);
            const int gcol = m0 + colL;
            const int rbase = mrow * 16 + ((lane >> 4) ? 8 : 0);
#pragma unroll
            for (int r = 0; r < 8; ++r) {
                int row = rbase + r;
                float xv = (sacc[r] - mu) * rsg;
                float e = ((l0 + row) == gcol) ? 0.0f : __expf(xv);
                Ps[row * 130 + colL] = f2bf(e);
            }
        }
        __syncthreads();

        // ---- row sums (packed 32-bit reads, unpack two bf16) ----
        {
            int row = t & 63, seg = t >> 6;
            const unsigned int* pp = (const unsigned int*)&Ps[row * 130 + seg * 32];
            float s = 0.f;
#pragma unroll
            for (int i = 0; i < 16; ++i) {
                unsigned int u = pp[i];
                s += __uint_as_float(u << 16) + __uint_as_float(u & 0xffff0000u);
            }
            atomicAdd(&rowsum[row], s);
        }

        // ---- out += P @ Vᵀ (each wave takes its K half of this tile) ----
#pragma unroll
        for (int ks = 0; ks < 2; ++ks) {
            int kk = (nh * 2 + ks) * 32;
            v16bf pa = load_fragA(Ps + (mrow * 16) * 130 + kk, 130, lane);
#pragma unroll
            for (int nd = 0; nd < 2; ++nd) {
                v16bf vb = load_fragB(Vs + (nd * 16) * 130 + kk, 130, lane);
                oacc[nd] = wmma_bf16(pa, vb, oacc[nd]);
            }
        }
    }
    __syncthreads();

    // combine nh=0 / nh=1 partial accumulators in LDS
    {
        const int rbase = mrow * 16 + ((lane >> 4) ? 8 : 0);
        const int col = lane & 15;
        if (nh == 0) {
#pragma unroll
            for (int nd = 0; nd < 2; ++nd)
#pragma unroll
                for (int r = 0; r < 8; ++r)
                    Ored[(rbase + r) * 33 + nd * 16 + col] = oacc[nd][r];
        }
        __syncthreads();
        if (nh == 1) {
#pragma unroll
            for (int nd = 0; nd < 2; ++nd)
#pragma unroll
                for (int r = 0; r < 8; ++r)
                    Ored[(rbase + r) * 33 + nd * 16 + col] += oacc[nd][r];
        }
        __syncthreads();
    }
    // normalize + write out[n, h*32+d, l0+row]
    for (int i = t; i < 64 * 32; i += 256) {
        int row = i >> 5, d = i & 31;
        float s = rowsum[row];
        float val = Ored[row * 33 + d];
        val = (s > 0.0f) ? (val / s) : 0.0f;      // NaN row -> 0 (matches ref)
        out[((size_t)n * 256 + h * 32 + d) * LG + l0 + row] = val;
    }
}

// ---------------------------------------------------------------------------
// Launch: full FeatureAlignmentLayer pipeline
// ---------------------------------------------------------------------------
extern "C" void kernel_launch(void* const* d_in, const int* in_sizes, int n_in,
                              void* d_out, int out_size, void* d_ws, size_t ws_size,
                              hipStream_t stream) {
    const float* f    = (const float*)d_in[0];
    const float* w_z  = (const float*)d_in[1];
    const float* gz_w = (const float*)d_in[2];
    const float* gz_b = (const float*)d_in[3];
    const float* w_qkv= (const float*)d_in[4];
    const float* gq_w = (const float*)d_in[5];
    const float* gq_b = (const float*)d_in[6];
    const float* w_out= (const float*)d_in[7];
    const float* go_w = (const float*)d_in[8];
    const float* go_b = (const float*)d_in[9];
    const float* w_f1 = (const float*)d_in[10];
    const float* g1_w = (const float*)d_in[11];
    const float* g1_b = (const float*)d_in[12];
    const float* w_f2 = (const float*)d_in[13];
    const float* g2_w = (const float*)d_in[14];
    const float* g2_b = (const float*)d_in[15];
    float* out = (float*)d_out;

    char* ws = (char*)d_ws;
    const size_t MB = 1024ull * 1024ull;
    float* bufA = (float*)(ws);              // 16 MB (up to 512 channels)
    float* bufB = (float*)(ws + 16 * MB);    // 8 MB
    float* bufC = (float*)(ws + 24 * MB);    // 16 MB (qkv)
    float* bufD = (float*)(ws + 40 * MB);    // 8 MB (attention output)
    float* bufE = (float*)(ws + 48 * MB);    // 8 MB (f2 residual)
    float* st   = (float*)(ws + 56 * MB);    // stats
    float *muA = st +  0, *rsA = st +  8;
    float *muB = st + 16, *rsB = st + 24;
    float *muC = st + 32, *rsC = st + 40;
    float *muD = st + 48, *rsD = st + 56;
    float *muE = st + 64, *rsE = st + 72;
    float *muF = st + 80, *rsF = st + 88;
    float *muH = st + 96, *rsH = st + 160;   // 64 each

    const int NCL  = 8 * 256 * 1024;   // 2,097,152
    const int NCL2 = 2 * NCL;

    // z = gelu(GN(w_z @ gelu(f)))
    k_gelu<<<NCL / 256, 256, 0, stream>>>(f, bufA, NCL);
    k_gemm<<<dim3(8, 4, 8), 256, 0, stream>>>(bufA, w_z, bufB, 256, 256);
    k_stats<<<8, 256, 0, stream>>>(bufB, muA, rsA, 256 * 1024);
    k_gn_apply<<<NCL / 256, 256, 0, stream>>>(bufB, nullptr, gz_w, gz_b, muA, rsA, bufA, 256, 1);
    // qkv = GN(w_qkv @ z)
    k_gemm<<<dim3(8, 8, 8), 256, 0, stream>>>(bufA, w_qkv, bufC, 256, 512);
    k_stats<<<8, 256, 0, stream>>>(bufC, muB, rsB, 512 * 1024);
    k_gn_apply<<<NCL2 / 256, 256, 0, stream>>>(bufC, nullptr, gq_w, gq_b, muB, rsB, bufC, 512, 0);
    // attention (LN stats analytically, then fused flash-style pass)
    k_attn_stats<<<64, 256, 0, stream>>>(bufC, muH, rsH);
    k_attn_fused<<<dim3(16, 64), 256, 0, stream>>>(bufC, muH, rsH, bufD);
    // out = gelu(LN(out))
    k_stats<<<8, 256, 0, stream>>>(bufD, muC, rsC, 256 * 1024);
    k_gn_apply<<<NCL / 256, 256, 0, stream>>>(bufD, nullptr, nullptr, nullptr, muC, rsC, bufA, 256, 1);
    // f2 = gelu(f + GN(w_out @ out))
    k_gemm<<<dim3(8, 4, 8), 256, 0, stream>>>(bufA, w_out, bufB, 256, 256);
    k_stats<<<8, 256, 0, stream>>>(bufB, muD, rsD, 256 * 1024);
    k_gn_apply<<<NCL / 256, 256, 0, stream>>>(bufB, f, go_w, go_b, muD, rsD, bufE, 256, 1);
    // h = gelu(GN(w_ffn1 @ f2))
    k_gemm<<<dim3(8, 8, 8), 256, 0, stream>>>(bufE, w_f1, bufC, 256, 512);
    k_stats<<<8, 256, 0, stream>>>(bufC, muE, rsE, 512 * 1024);
    k_gn_apply<<<NCL2 / 256, 256, 0, stream>>>(bufC, nullptr, g1_w, g1_b, muE, rsE, bufA, 512, 1);
    // out = gelu(f2 + GN(w_ffn2 @ h))
    k_gemm<<<dim3(8, 4, 8), 256, 0, stream>>>(bufA, w_f2, bufB, 512, 256);
    k_stats<<<8, 256, 0, stream>>>(bufB, muF, rsF, 256 * 1024);
    k_gn_apply<<<NCL / 256, 256, 0, stream>>>(bufB, bufE, g2_w, g2_b, muF, rsF, out, 256, 1);
}
